// FacetCrossAttention_66176856097396
// MI455X (gfx1250) — matrix operand order repaired and checked by reading, not compiled, not run on hardware
//
#include <hip/hip_runtime.h>
#include <math.h>

#define Dm    1024
#define Hn    16
#define DHn   64
#define Bn    8
#define Pn    512
#define Kn    16
#define BPn   4096
#define CHUNK 1024
#define NCHUNK 4
#define GSTR  1032   // padded LDS row stride for gene tile (f16 elems)

typedef __attribute__((ext_vector_type(16))) _Float16 v16h;
typedef __attribute__((ext_vector_type(8)))  _Float16 v8h;
typedef __attribute__((ext_vector_type(8)))  float    v8f;

// ---- WMMA fragment loaders (f16, 16x16x32) -------------------------------
// A-matrix 16x32: lane = row (mod 16); element e -> K = e + (e>=8?8:0) + (laneHi?8:0)
// i.e. elems 0..7 = p[0..7], elems 8..15 = p[16..23], with p = base + c0 + laneHi*8
__device__ __forceinline__ v16h ld_a_frag(const _Float16* p) {
  v8h lo = *(const v8h*)(p);
  v8h hi = *(const v8h*)(p + 16);
  v16h r;
#pragma unroll
  for (int i = 0; i < 8; ++i) { r[i] = lo[i]; r[i + 8] = hi[i]; }
  return r;
}
// B-matrix 32x16: lane = column (mod 16); element e -> K = e + (e>=8?8:0) + (laneHi?8:0)
// i.e. elems 0..7 = p[0..7], elems 8..15 = p[16..23], with p = base + c0 + laneHi*8
__device__ __forceinline__ v16h ld_b_frag(const _Float16* p) {
  v8h lo = *(const v8h*)(p);
  v8h hi = *(const v8h*)(p + 16);
  v16h r;
#pragma unroll
  for (int i = 0; i < 8; ++i) { r[i] = lo[i]; r[i + 8] = hi[i]; }
  return r;
}
__device__ __forceinline__ v8f wmma16(v16h a, v16h b, v8f c) {
  return __builtin_amdgcn_wmma_f32_16x16x32_f16(false, a, false, b, (short)0, c,
                                                false, false);
}

// ---- Kernel 0: transpose 1024x1024 fp32 -> f16 (dst[d][c] = src[c][d]) ----
__global__ __launch_bounds__(256) void k0_transpose(const float* __restrict__ src,
                                                    _Float16* __restrict__ dst) {
  __shared__ float t[32][33];
  int bx = blockIdx.x * 32;            // src column block (d)
  int by = blockIdx.y * 32;            // src row block (c)
  int lx = threadIdx.x & 31, ly = threadIdx.x >> 5;
#pragma unroll
  for (int i = 0; i < 32; i += 8)
    t[ly + i][lx] = src[(size_t)(by + ly + i) * Dm + bx + lx];
  __syncthreads();
#pragma unroll
  for (int i = 0; i < 32; i += 8)
    dst[(size_t)(bx + ly + i) * Dm + by + lx] = (_Float16)t[lx][ly + i];
}

// ---- Kernel 1a: Qflat = cell_query @ wq + bq  (8 x 1024) ------------------
__global__ __launch_bounds__(256) void k1a_qproj(const float* __restrict__ cq,
                                                 const float* __restrict__ wq,
                                                 const float* __restrict__ bq,
                                                 float* __restrict__ Qflat) {
  int b = blockIdx.x, tid = threadIdx.x;
  __shared__ float sx[Dm];
  for (int i = tid; i < Dm; i += 256) sx[i] = cq[b * Dm + i];
  __syncthreads();
  for (int d = tid; d < Dm; d += 256) {
    float acc = bq[d];
    for (int c = 0; c < Dm; ++c) acc += sx[c] * wq[(size_t)c * Dm + d];
    Qflat[b * Dm + d] = acc;
  }
}

// ---- Kernel 1b: Qk[b,h,c] = (1/8)*sum_dh wk[c,h*64+dh]*Qflat[b,h*64+dh] ---
//      also qkbias[b,h] = (1/8)*Qflat[b,h-slice] . bk[h-slice]
__global__ __launch_bounds__(256) void k1b_qk(const float* __restrict__ wk,
                                              const float* __restrict__ bk,
                                              const float* __restrict__ Qflat,
                                              _Float16* __restrict__ qk,
                                              float* __restrict__ qkbias) {
  int h = blockIdx.x, b = blockIdx.y, tid = threadIdx.x;
  __shared__ float sq[DHn];
  if (tid < DHn) sq[tid] = Qflat[b * Dm + h * DHn + tid];
  __syncthreads();
  if (tid == 0) {
    float s = 0.f;
    for (int d = 0; d < DHn; ++d) s += sq[d] * bk[h * DHn + d];
    qkbias[b * Hn + h] = 0.125f * s;
  }
  for (int c = tid; c < Dm; c += 256) {
    const float* w = wk + (size_t)c * Dm + h * DHn;
    float acc = 0.f;
#pragma unroll 8
    for (int d = 0; d < DHn; ++d) acc += w[d] * sq[d];
    qk[((size_t)b * Hn + h) * Dm + c] = (_Float16)(acc * 0.125f);
  }
}

// ---- Kernel 2: per (b,p): scores (WMMA) -> sparsemax -> y = alpha^T gene ---
__global__ __launch_bounds__(256) void k2_attn(const float* __restrict__ gene,
                                               const float* __restrict__ conf,
                                               const _Float16* __restrict__ qk,
                                               const float* __restrict__ qkbias,
                                               _Float16* __restrict__ Y,  // (H,CHUNK,D)
                                               float* __restrict__ alpha_out,
                                               int bp_base) {
  __shared__ _Float16 sgene[Kn * GSTR];
  __shared__ float spart[8][16][16];
  __shared__ float sscore[16][16];
  __shared__ float salpha[16][16];
  __shared__ float slconf[16];

  int lbp = blockIdx.x;                 // local row in chunk
  int bp  = bp_base + lbp;              // global (b*P+p)
  int b   = bp >> 9;                    // P = 512
  int tid = threadIdx.x, lane = tid & 31, wave = tid >> 5;
  int hi  = (lane >> 4) & 1, l15 = lane & 15;

  // stage gene rows (K=16 x 1024) as f16 in LDS
  const float* g = gene + (size_t)bp * (Kn * Dm);
  for (int i = tid; i < Kn * Dm; i += 256) {
    int k = i >> 10, c = i & 1023;
    sgene[k * GSTR + c] = (_Float16)g[i];
  }
  if (tid < Kn) slconf[tid] = logf(fmaxf(conf[(size_t)bp * Kn + tid], 1e-6f));
  __syncthreads();

  // scores[h,k]: A = Qk[b] (16 x 1024), B = gene^T; each wave does 128 of K=1024
  v8f acc = {};
  const _Float16* aBase = qk + ((size_t)b * Hn + l15) * Dm;
  const _Float16* gRow  = &sgene[l15 * GSTR];
#pragma unroll
  for (int j = 0; j < 4; ++j) {
    int c0 = wave * 128 + j * 32;
    v16h a  = ld_a_frag(aBase + c0 + hi * 8);
    v16h bb = ld_b_frag(gRow + c0 + hi * 8);
    acc = wmma16(a, bb, acc);
  }
#pragma unroll
  for (int r = 0; r < 8; ++r) spart[wave][r + hi * 8][l15] = acc[r];
  __syncthreads();
  {
    int h = tid >> 4, k = tid & 15;   // 256 threads == 16x16
    float s = 0.f;
#pragma unroll
    for (int w = 0; w < 8; ++w) s += spart[w][h][k];
    sscore[h][k] = s + slconf[k] + qkbias[b * Hn + h];
  }
  __syncthreads();

  // sparsemax over k (one thread per head)
  if (tid < Hn) {
    int h = tid;
    float z[16];
    for (int k = 0; k < 16; ++k) z[k] = sscore[h][k];
    for (int i = 1; i < 16; ++i) {   // insertion sort, descending
      float key = z[i]; int j = i - 1;
      for (; j >= 0 && z[j] < key; --j) z[j + 1] = z[j];
      z[j + 1] = key;
    }
    float cum = 0.f, csel = z[0]; int ksup = 1;
    for (int r = 1; r <= 16; ++r) {
      cum += z[r - 1];
      if (1.0f + (float)r * z[r - 1] > cum) { ksup = r; csel = cum; }
    }
    float tau = (csel - 1.0f) / (float)ksup;
    for (int k = 0; k < 16; ++k) {
      float a = fmaxf(sscore[h][k] - tau, 0.0f);
      salpha[h][k] = a;
      alpha_out[(size_t)bp * (Hn * Kn) + h * Kn + k] = a;
    }
  }
  __syncthreads();

  // y[h,c] = sum_k alpha[h,k] * gene[k,c]   -> Y[h][lbp][c] (f16)
  for (int c = tid; c < Dm; c += 256) {
    float gk[16];
#pragma unroll
    for (int k = 0; k < 16; ++k) gk[k] = (float)sgene[k * GSTR + c];
#pragma unroll
    for (int h = 0; h < Hn; ++h) {
      float y = 0.f;
#pragma unroll
      for (int k = 0; k < 16; ++k) y += salpha[h][k] * gk[k];
      Y[((size_t)h * CHUNK + lbp) * Dm + c] = (_Float16)y;
    }
  }
}

// ---- Kernel 3: ctx[:, h*64:(h+1)*64] = Y[h] @ wv[:, h-slice] + bv ---------
__global__ __launch_bounds__(256) void k3_ctx(const _Float16* __restrict__ Y,
                                              const _Float16* __restrict__ wvT,
                                              const float* __restrict__ bv,
                                              _Float16* __restrict__ ctx) {
  int h  = blockIdx.y;
  int mb = blockIdx.x * 64;                       // local row tile base
  int tid = threadIdx.x, lane = tid & 31, wave = tid >> 5;
  int hi = (lane >> 4) & 1, l15 = lane & 15;
  int tm = wave >> 1;                              // tile row 0..3
  int tn0 = (wave & 1) * 2, tn1 = tn0 + 1;         // tile cols
  int row = mb + tm * 16 + l15;
  const _Float16* aBase  = Y + ((size_t)h * CHUNK + row) * Dm;
  const _Float16* b0Base = wvT + (size_t)(h * DHn + tn0 * 16 + l15) * Dm;
  const _Float16* b1Base = wvT + (size_t)(h * DHn + tn1 * 16 + l15) * Dm;
  v8f acc0 = {}, acc1 = {};
  for (int c0 = 0; c0 < Dm; c0 += 32) {
    v16h a  = ld_a_frag(aBase + c0 + hi * 8);
    v16h b0 = ld_b_frag(b0Base + c0 + hi * 8);
    v16h b1 = ld_b_frag(b1Base + c0 + hi * 8);
    acc0 = wmma16(a, b0, acc0);
    acc1 = wmma16(a, b1, acc1);
  }
  int c0g = h * DHn + tn0 * 16 + l15;
  int c1g = h * DHn + tn1 * 16 + l15;
  float bv0 = bv[c0g], bv1 = bv[c1g];   // sparsemax weights sum to 1 -> +bv
#pragma unroll
  for (int r = 0; r < 8; ++r) {
    int rr = mb + tm * 16 + r + hi * 8;
    ctx[(size_t)rr * Dm + c0g] = (_Float16)(acc0[r] + bv0);
    ctx[(size_t)rr * Dm + c1g] = (_Float16)(acc1[r] + bv1);
  }
}

// ---- Kernel 4: out = LayerNorm(ctx @ wo + bo) * gamma + beta --------------
__global__ __launch_bounds__(256) void k4_out(const _Float16* __restrict__ ctx,
                                              const _Float16* __restrict__ woT,
                                              const float* __restrict__ bo,
                                              const float* __restrict__ gamma,
                                              const float* __restrict__ beta,
                                              float* __restrict__ out) {
  __shared__ float sredS[16][8][16];
  __shared__ float sredQ[16][8][16];
  __shared__ float smu[16], srv[16];
  int mb = blockIdx.x * 16;                        // local row tile
  int tid = threadIdx.x, lane = tid & 31, wave = tid >> 5;
  int hi = (lane >> 4) & 1, l15 = lane & 15;
  const _Float16* aBase = ctx + (size_t)(mb + l15) * Dm;
  v8f acc[8];
#pragma unroll
  for (int j = 0; j < 8; ++j) acc[j] = (v8f){};
  for (int c0 = 0; c0 < Dm; c0 += 32) {
    v16h a = ld_a_frag(aBase + c0 + hi * 8);
#pragma unroll
    for (int j = 0; j < 8; ++j) {
      int n = wave * 128 + j * 16 + l15;
      v16h bb = ld_b_frag(woT + (size_t)n * Dm + c0 + hi * 8);
      acc[j] = wmma16(a, bb, acc[j]);
    }
  }
  // add bias, per-lane partial LN stats
  float ps[8], ps2[8];
#pragma unroll
  for (int r = 0; r < 8; ++r) { ps[r] = 0.f; ps2[r] = 0.f; }
#pragma unroll
  for (int j = 0; j < 8; ++j) {
    float bj = bo[wave * 128 + j * 16 + l15];
#pragma unroll
    for (int r = 0; r < 8; ++r) {
      float v = acc[j][r] + bj;
      acc[j][r] = v;
      ps[r] += v;
      ps2[r] += v * v;
    }
  }
#pragma unroll
  for (int r = 0; r < 8; ++r) {
    sredS[r + hi * 8][wave][l15] = ps[r];
    sredQ[r + hi * 8][wave][l15] = ps2[r];
  }
  __syncthreads();
  if (tid < 16) {
    float s = 0.f, s2 = 0.f;
    for (int w = 0; w < 8; ++w)
      for (int l = 0; l < 16; ++l) { s += sredS[tid][w][l]; s2 += sredQ[tid][w][l]; }
    float mu = s * (1.0f / 1024.0f);
    float var = s2 * (1.0f / 1024.0f) - mu * mu;
    smu[tid] = mu;
    srv[tid] = rsqrtf(var + 1e-5f);
  }
  __syncthreads();
#pragma unroll
  for (int j = 0; j < 8; ++j) {
    int col = wave * 128 + j * 16 + l15;
    float gm = gamma[col], bt = beta[col];
#pragma unroll
    for (int r = 0; r < 8; ++r) {
      int row = r + hi * 8;
      out[(size_t)(mb + row) * Dm + col] = (acc[j][r] - smu[row]) * srv[row] * gm + bt;
    }
  }
}

// ---- host launch ----------------------------------------------------------
extern "C" void kernel_launch(void* const* d_in, const int* in_sizes, int n_in,
                              void* d_out, int out_size, void* d_ws, size_t ws_size,
                              hipStream_t stream) {
  const float* cq    = (const float*)d_in[0];
  const float* gene  = (const float*)d_in[1];
  const float* conf  = (const float*)d_in[2];
  const float* wq    = (const float*)d_in[3];
  const float* bq    = (const float*)d_in[4];
  const float* wk    = (const float*)d_in[5];
  const float* bk    = (const float*)d_in[6];
  const float* wv    = (const float*)d_in[7];
  const float* bv    = (const float*)d_in[8];
  const float* wo    = (const float*)d_in[9];
  const float* bo    = (const float*)d_in[10];
  const float* gamma = (const float*)d_in[11];
  const float* beta  = (const float*)d_in[12];

  float* out       = (float*)d_out;                 // (B,P,D)
  float* alpha_out = out + (size_t)BPn * Dm;        // (B,P,H,K)

  char* ws = (char*)d_ws;
  float*     Qflat  = (float*)(ws + 0);                       //  32 KB
  float*     qkbias = (float*)(ws + 32768);                   //   4 KB slot
  _Float16*  qk     = (_Float16*)(ws + 36864);                // 256 KB
  _Float16*  wvT    = (_Float16*)(ws + 299008);               //   2 MB
  _Float16*  woT    = (_Float16*)(ws + 2396160);              //   2 MB
  _Float16*  ctx    = (_Float16*)(ws + 4493312);              //   2 MB (chunk)
  _Float16*  Ybuf   = (_Float16*)(ws + 6590464);              //  32 MB (chunk)
  // total ~40.1 MB of d_ws used

  k0_transpose<<<dim3(32, 32), 256, 0, stream>>>(wv, wvT);
  k0_transpose<<<dim3(32, 32), 256, 0, stream>>>(wo, woT);
  k1a_qproj<<<Bn, 256, 0, stream>>>(cq, wq, bq, Qflat);
  k1b_qk<<<dim3(Hn, Bn), 256, 0, stream>>>(wk, bk, Qflat, qk, qkbias);

  for (int chunk = 0; chunk < NCHUNK; ++chunk) {
    int bp_base = chunk * CHUNK;
    k2_attn<<<CHUNK, 256, 0, stream>>>(gene, conf, qk, qkbias, Ybuf, alpha_out,
                                       bp_base);
    k3_ctx<<<dim3(CHUNK / 64, Hn), 256, 0, stream>>>(Ybuf, wvT, bv, ctx);
    k4_out<<<CHUNK / 16, 256, 0, stream>>>(ctx, woT, bo, gamma, beta,
                                           out + (size_t)bp_base * Dm);
  }
}
